// BlocksparseConv_3229815406650
// MI455X (gfx1250) — matrix operand order repaired and checked by reading, not compiled, not run on hardware
//
#include <hip/hip_runtime.h>
#include <hip/hip_bf16.h>

// Block-sparse 3x3 conv, NCHW f32 -> f32, implicit GEMM with
// v_wmma_f32_16x16x32_f16 (f16 inputs, f32 accumulate).
//
// B=16, CIN=COUT=128, H=W=128, KH=KW=3, pad=1, 32x32 channel-block mask.
//
// Workgroup = 256 threads = 8 waves. Each workgroup computes a
// 128(cout) x 4(h) x 16(w) output tile; wave t owns couts [16t,16t+16)
// for all 4 rows (4 accumulator fragments, A fragments reused 4x).

#define BATCH 16
#define CIN   128
#define COUT  128
#define HH    128
#define WW    128
#define BLK   32           // mask block size over (cout, cin)
#define NOBK  4            // COUT / BLK
#define NIBK  4            // CIN / BLK
#define NT    8            // COUT / 16  (cout tiles of 16)
#define HTILE 4            // output rows per workgroup
#define HALO  (HTILE + 2)  // input halo rows

// LDS input tile: 6 halo rows x 18 halo cols x 128 cin (f16), cin innermost.
// cin-stride padded 128 -> 136 halfs (272 B) so the 16 lanes of a B-fragment
// ds_load_b128 (addresses 272 B apart -> dword-bank stride 68 -> bank 4n)
// hit distinct banks within each half-wave.
#define CSTRIDE 136

typedef __attribute__((ext_vector_type(16))) _Float16 v16h;
typedef __attribute__((ext_vector_type(8)))  float    v8f;

// ---------------------------------------------------------------------------
// Prelude 1: per-(obk,ibk) activity flag. Mask is constant inside each 32x32
// block, so one element per block suffices.
// ---------------------------------------------------------------------------
__global__ void bsconv_flags_kernel(const float* __restrict__ mask,
                                    int* __restrict__ flags) {
    int tid = threadIdx.x;
    if (tid < NOBK * NIBK) {
        int obk = tid >> 2, ibk = tid & 3;
        float m = mask[(size_t)(obk * BLK) * (CIN * 9) + (size_t)(ibk * BLK) * 9];
        flags[tid] = (m != 0.0f) ? 1 : 0;
    }
}

// ---------------------------------------------------------------------------
// Prelude 2: pack weight*mask into f16 A-fragments in this kernel's contiguous-k
// order (not the ISA table order; B loads use the same), 16x32 MxK:
//   lane l: m = l&15, hi = l>>4
//   half j (0..15): k = hi*16 + j
// One fragment per (cout_tile t, ibk, tap): stored lane-contiguous so the
// main kernel reads 32 B per lane (2x global_load_b128, coalesced, L2-hot).
// ---------------------------------------------------------------------------
__global__ void bsconv_pack_kernel(const float* __restrict__ weight,
                                   const float* __restrict__ mask,
                                   _Float16* __restrict__ wpack) {
    int idx = blockIdx.x * 256 + threadIdx.x;
    if (idx >= NT * NIBK * 9 * 512) return;
    int frag   = idx >> 9;
    int within = idx & 511;
    int lane = within >> 4;
    int j    = within & 15;
    int t    = frag / (NIBK * 9);
    int rem  = frag % (NIBK * 9);
    int ibk  = rem / 9;
    int tap  = rem % 9;
    int kh = tap / 3, kw = tap % 3;
    int m  = lane & 15;
    int hi = lane >> 4;
    int k  = hi * 16 + j;
    int cout = t * 16 + m;
    int cin  = ibk * BLK + k;
    size_t widx = (((size_t)cout * CIN + cin) * 3 + kh) * 3 + kw;
    wpack[idx] = (_Float16)(weight[widx] * mask[widx]);
}

// ---------------------------------------------------------------------------
// Main kernel. Grid: (WW/16, HH/HTILE, BATCH). Block: 256 threads = 8 waves.
// ---------------------------------------------------------------------------
__global__ void __launch_bounds__(256)
bsconv_main_kernel(const float* __restrict__ x,
                   const float* __restrict__ bias,
                   const int* __restrict__ flags,
                   const _Float16* __restrict__ wpack,
                   float* __restrict__ out) {
    __shared__ __attribute__((aligned(16))) _Float16 lds_x[HALO * 18 * CSTRIDE];

    const int tid  = threadIdx.x;
    const int lane = tid & 31;
    const int wave = tid >> 5;           // cout tile t = 0..7
    const int w0   = blockIdx.x * 16;
    const int h0   = blockIdx.y * HTILE;
    const int b    = blockIdx.z;

    // ---- Stage input halo tile into LDS as f16 (cin innermost) ----
    // One (row, cin) segment per wave-iteration; LANES span the contiguous
    // w direction (lane = halo column) so each global_load_b32 touches at
    // most 2 cache lines instead of 32.
    #pragma unroll 4
    for (int seg = wave; seg < HALO * CIN; seg += NT) {
        int rr  = seg >> 7;              // 0..5 halo row
        int cin = seg & 127;
        int hr  = h0 - 1 + rr;
        int wc  = w0 - 1 + lane;         // halo column = lane
        float v = 0.0f;
        if (lane < 18 && hr >= 0 && hr < HH && wc >= 0 && wc < WW)
            v = x[(((size_t)b * CIN + cin) * HH + hr) * WW + wc];
        if (lane < 18)
            lds_x[(rr * 18 + lane) * CSTRIDE + cin] = (_Float16)v;
    }
    __syncthreads();

    // ---- WMMA compute: 4 accumulator fragments (one per output row) ----
    const int obk = wave >> 1;           // 32-wide mask block row
    const int hi  = lane >> 4;           // half-wave select
    const int n   = lane & 15;           // output pixel within tile

    v8f acc[HTILE];
    #pragma unroll
    for (int ht = 0; ht < HTILE; ++ht) acc[ht] = (v8f){};

    for (int ibk = 0; ibk < NIBK; ++ibk) {
        int act = __builtin_amdgcn_readfirstlane(flags[obk * NIBK + ibk]);
        if (!act) continue;
        #pragma unroll
        for (int kh = 0; kh < 3; ++kh) {
            #pragma unroll
            for (int kw = 0; kw < 3; ++kw) {
                const int frag = (wave * NIBK + ibk) * 9 + kh * 3 + kw;
                // A fragment: pre-packed, lane-contiguous 32 B; reused for
                // all 4 output rows.
                v16h a = *(const v16h*)(wpack + (size_t)frag * 512 + lane * 16);
                const int cbase = ibk * BLK + hi * 16;
                #pragma unroll
                for (int ht = 0; ht < HTILE; ++ht) {
                    // B fragment (32x16 KxN): lane holds pixel n, halfs
                    // j=0..15 cover cin = cbase + j (contiguous in LDS).
                    const _Float16* bsrc =
                        &lds_x[((ht + kh) * 18 + n + kw) * CSTRIDE + cbase];
                    v16h bm = *(const v16h*)bsrc;
                    acc[ht] = __builtin_amdgcn_wmma_f32_16x16x32_f16(
                        false, a, false, bm, (short)0, acc[ht], false, false);
                }
            }
        }
    }

    // ---- Bias + store (C/D layout: VGPR r -> M = r + 8*hi, lane n -> N) ----
    #pragma unroll
    for (int r = 0; r < 8; ++r) {
        int cout = wave * 16 + r + 8 * hi;
        float bb = bias[cout];
        float* orow = out + (((size_t)b * COUT + cout) * HH + h0) * WW + w0 + n;
        #pragma unroll
        for (int ht = 0; ht < HTILE; ++ht)
            orow[(size_t)ht * WW] = acc[ht][r] + bb;
    }
}

// ---------------------------------------------------------------------------
// Launch wrapper.
// Inputs (setup_inputs order): x[f32], weight[f32], bias[f32], mask[f32].
// d_ws layout: [0,64)   int flags[16]
//              [1024, 1024 + 8*4*9*512*2) packed f16 weights (~288 KB)
// ---------------------------------------------------------------------------
extern "C" void kernel_launch(void* const* d_in, const int* in_sizes, int n_in,
                              void* d_out, int out_size, void* d_ws, size_t ws_size,
                              hipStream_t stream) {
    const float* x      = (const float*)d_in[0];
    const float* weight = (const float*)d_in[1];
    const float* bias   = (const float*)d_in[2];
    const float* mask   = (const float*)d_in[3];
    float* out          = (float*)d_out;

    int* flags      = (int*)d_ws;
    _Float16* wpack = (_Float16*)((char*)d_ws + 1024);

    bsconv_flags_kernel<<<1, 32, 0, stream>>>(mask, flags);

    const int pack_elems = NT * NIBK * 9 * 512;   // 147456
    bsconv_pack_kernel<<<(pack_elems + 255) / 256, 256, 0, stream>>>(
        weight, mask, wpack);

    dim3 grid(WW / 16, HH / HTILE, BATCH);
    bsconv_main_kernel<<<grid, 256, 0, stream>>>(x, bias, flags, wpack, out);
}
